// TokenAttnDecoderRNN_50345606644386
// MI455X (gfx1250) — matrix operand order repaired and checked by reading, not compiled, not run on hardware
//
#include <hip/hip_runtime.h>
#include <cstddef>

#define BB 64
#define HH 1024
#define HMM 512
#define EE 512
#define VV 32000
#define CATW 3584   // E + 2H + 2HM
#define MT 128      // attention M-tile rows per workgroup

typedef __attribute__((ext_vector_type(16))) __bf16 v16bf;
typedef __attribute__((ext_vector_type(8)))  float  v8f;
typedef __attribute__((ext_vector_type(2)))  float  v2f;

__device__ __forceinline__ float fast_tanh(float x) {
    x = fminf(fmaxf(x, -15.f), 15.f);
    float t = __expf(2.f * x);
    return (t - 1.f) / (t + 1.f);
}

// ---------------------------------------------------------------- embedding
__global__ void embed_kernel(const int* __restrict__ word,
                             const float* __restrict__ emb,
                             float* __restrict__ out_we,
                             float* __restrict__ cat) {
    int b = blockIdx.x;
    const float* row = emb + (size_t)word[b] * EE;
    for (int i = threadIdx.x; i < EE; i += blockDim.x) {
        float v = row[i];
        out_we[(size_t)b * EE + i] = v;
        cat[(size_t)b * CATW + i]  = v;
    }
}

// -------------------------------------------------- generic small fp32 GEMM
// C[r*ldc+n] = act( bias[n] + sum_k A[r*lda+k] * W[n*ldw+woff+k] )
__global__ void gemm_rowdot(const float* __restrict__ A, int lda,
                            const float* __restrict__ W, int ldw, int woff,
                            const float* __restrict__ bias,
                            float* __restrict__ C, int ldc,
                            int R, int N, int K, int act) {
    int idx = blockIdx.x * blockDim.x + threadIdx.x;
    if (idx >= R * N) return;
    int r = idx / N, n = idx % N;
    const float4* a = (const float4*)(A + (size_t)r * lda);
    const float4* w = (const float4*)(W + (size_t)n * ldw + woff);
    float acc = 0.f;
    int K4 = K >> 2;
    for (int k = 0; k < K4; ++k) {
        float4 av = a[k], wv = w[k];
        acc += av.x * wv.x + av.y * wv.y + av.z * wv.z + av.w * wv.w;
    }
    acc += bias[n];
    if (act) acc = fmaxf(acc, 0.f);
    C[(size_t)r * ldc + n] = acc;
}

// ------------------------------------------- convert W[:, off:off+K] -> bf16
__global__ void cvt_we_kernel(const float* __restrict__ W, int ldw, int off,
                              __bf16* __restrict__ dst, int N, int K) {
    int idx = blockIdx.x * blockDim.x + threadIdx.x;
    if (idx >= N * K) return;
    int n = idx / K, k = idx % K;
    dst[(size_t)n * K + k] = (__bf16)W[(size_t)n * ldw + off + k];
}

// ------------------------------------------------- attention score (bf16 WMMA)
// scores[r] = sum_n v[n] * tanh( sum_k X[r,k]*We[n,k] + hWb[r%B, n] )
// X = enc in (T,B,D) layout => row r = t*B+b contiguous in D.
// Block: 256 threads, MT=128 rows. Enc tile staged fp32->bf16 in LDS; each
// W_e fragment load feeds 8 WMMAs (8 M-subtiles) -> 8x less L2 traffic on W_e.
template <int D>
__global__ void attn_score_kernel(const float* __restrict__ X,
                                  const __bf16* __restrict__ We,
                                  const float* __restrict__ hWb,
                                  const float* __restrict__ vvec,
                                  float* __restrict__ scores) {
    constexpr int NSUB = D / 128;                 // 16-wide n-tiles per wave
    extern __shared__ char smem[];
    __bf16* xs    = (__bf16*)smem;                // MT * D bf16
    float*  wpart = (float*)(smem + (size_t)MT * D * 2); // 8 * MT floats

    int tid  = threadIdx.x;
    int lane = tid & 31;
    int wave = tid >> 5;                          // 0..7
    int l15  = lane & 15;
    int grp  = lane >> 4;                         // 0/1
    int row_base = blockIdx.x * MT;

    // ---- stage enc tile: fp32 global -> bf16 LDS --------------------------
    {
        const float* src = X + (size_t)row_base * D;
        const int total = MT * D;
        for (int i = tid * 4; i < total; i += 256 * 4) {
            float4 v = *(const float4*)(src + i);
            union { __bf16 h[4]; uint2 u; } p;
            p.h[0] = (__bf16)v.x; p.h[1] = (__bf16)v.y;
            p.h[2] = (__bf16)v.z; p.h[3] = (__bf16)v.w;
            *(uint2*)(xs + i) = p.u;
        }
        for (int i = tid; i < 8 * MT; i += 256) wpart[i] = 0.f;
    }
    __syncthreads();

    v8f zero = {0.f, 0.f, 0.f, 0.f, 0.f, 0.f, 0.f, 0.f};

    for (int s = 0; s < NSUB; ++s) {
        int n = (wave * NSUB + s) * 16 + l15;
        const __bf16* wrow = We + (size_t)n * D;

        v8f acc[8];
#pragma unroll
        for (int mt = 0; mt < 8; ++mt) acc[mt] = zero;

        for (int k0 = 0; k0 < D; k0 += 32) {
            union { uint4 u[2]; v16bf v; } bf;
            const uint4* wq = (const uint4*)(wrow + k0 + 16 * grp);
            bf.u[0] = wq[0];
            bf.u[1] = wq[1];
#pragma unroll
            for (int mt = 0; mt < 8; ++mt) {
                const __bf16* ar = xs + (size_t)(mt * 16 + l15) * D + k0;
                union { uint4 u[2]; v16bf v; } af;
                af.u[0] = *(const uint4*)(ar + 16 * grp);       // k 16g..16g+7
                af.u[1] = *(const uint4*)(ar + 16 * grp + 8);   // k 16g+8..16g+15
                acc[mt] = __builtin_amdgcn_wmma_f32_16x16x32_bf16(
                    false, af.v, false, bf.v, (short)0, acc[mt], false, false);
            }
        }

        // epilogue for this n-subtile: add hWb, tanh, dot v, reduce over 16 n
        float vn = vvec[n];
#pragma unroll
        for (int mt = 0; mt < 8; ++mt) {
#pragma unroll
            for (int j = 0; j < 8; ++j) {
                int m = mt * 16 + j + 8 * grp;          // row within tile
                int b = (row_base + m) & (BB - 1);
                float e = acc[mt][j] + hWb[(size_t)b * D + n];
                float p = vn * fast_tanh(e);
#pragma unroll
                for (int o = 8; o >= 1; o >>= 1) p += __shfl_xor(p, o, 16);
                if (l15 == 0) wpart[wave * MT + m] += p;   // fixed lane/slot: deterministic
            }
        }
    }
    __syncthreads();
    if (tid < MT) {
        float ssum = 0.f;
        for (int w = 0; w < 8; ++w) ssum += wpart[w * MT + tid];
        scores[row_base + tid] = ssum;
    }
}

// --------------------------------------------- softmax over t + weighted ctx
__global__ void softmax_ctx_kernel(const float* __restrict__ enc,   // (T,B,D)
                                   const float* __restrict__ scores,// (T*B)
                                   float* __restrict__ dst, int dst_ld,
                                   int T, int D) {
    int b = blockIdx.x, tid = threadIdx.x;
    __shared__ float w[256];
    __shared__ float red[256];
    float s = (tid < T) ? scores[(size_t)tid * BB + b] : -3.4e38f;
    red[tid] = s; __syncthreads();
    for (int st = 128; st > 0; st >>= 1) {
        if (tid < st) red[tid] = fmaxf(red[tid], red[tid + st]);
        __syncthreads();
    }
    float mx = red[0]; __syncthreads();
    float e = (tid < T) ? __expf(s - mx) : 0.f;
    red[tid] = e; __syncthreads();
    for (int st = 128; st > 0; st >>= 1) {
        if (tid < st) red[tid] += red[tid + st];
        __syncthreads();
    }
    float inv = 1.f / red[0];
    w[tid] = e * inv;
    __syncthreads();
    for (int d = tid; d < D; d += 256) {
        float acc = 0.f;
        for (int t = 0; t < T; ++t)
            acc += w[t] * enc[((size_t)t * BB + b) * D + d];
        dst[(size_t)b * dst_ld + d] = acc;
    }
}

// ------------------------------------------------------------- GRU gate fuse
__global__ void gru_gate_kernel(const float* __restrict__ gi,
                                const float* __restrict__ gh,
                                const float* __restrict__ h,
                                float* __restrict__ hnew_ws,
                                float* __restrict__ hnew_out) {
    int idx = blockIdx.x * blockDim.x + threadIdx.x;
    if (idx >= BB * HH) return;
    int b = idx / HH, j = idx % HH;
    size_t o3 = (size_t)b * 3 * HH;
    float ir = gi[o3 + j],          hr = gh[o3 + j];
    float iz = gi[o3 + HH + j],     hz = gh[o3 + HH + j];
    float in_ = gi[o3 + 2*HH + j],  hn = gh[o3 + 2*HH + j];
    float r = 1.f / (1.f + __expf(-(ir + hr)));
    float z = 1.f / (1.f + __expf(-(iz + hz)));
    float n = fast_tanh(in_ + r * hn);
    float hv = (1.f - z) * n + z * h[(size_t)b * HH + j];
    hnew_ws[(size_t)b * HH + j]  = hv;
    hnew_out[(size_t)b * HH + j] = hv;
}

// ------------------------------------------------- logits via fp32 WMMA (K=4)
// out[m,n] = bias[n] + sum_k A[m,k]*W[n,k]   M=64, N=32000, K=1024
__global__ void logits_wmma_kernel(const float* __restrict__ A,
                                   const float* __restrict__ W,
                                   const float* __restrict__ bias,
                                   float* __restrict__ out) {
    const int K = HH;
    int lane = threadIdx.x & 31;
    int wave = threadIdx.x >> 5;
    int l15 = lane & 15, grp = lane >> 4;
    int n = blockIdx.x * 128 + wave * 16 + l15;

    v8f zero = {0.f, 0.f, 0.f, 0.f, 0.f, 0.f, 0.f, 0.f};
    v8f acc[4];
#pragma unroll
    for (int t = 0; t < 4; ++t) acc[t] = zero;

    const float* wrow = W + (size_t)n * K;
    for (int k0 = 0; k0 < K; k0 += 4) {
        if ((k0 & 127) == 0)
            __builtin_prefetch(wrow + k0 + 512, 0, 1);
        v2f bf;                     // B 4x16: lanes0-15 K=k0,k0+1; lanes16-31 K=k0+2,k0+3
        bf[0] = wrow[k0 + 2 * grp];
        bf[1] = wrow[k0 + 2 * grp + 1];
#pragma unroll
        for (int t = 0; t < 4; ++t) {
            int m = t * 16 + l15;
            const float* ar = A + (size_t)m * K + k0 + 2 * grp;
            v2f af;                 // A 16x4: v0=K(k0+2g), v1=K(k0+2g+1)
            af[0] = ar[0];
            af[1] = ar[1];
            acc[t] = __builtin_amdgcn_wmma_f32_16x16x4_f32(
                false, af, false, bf, (short)0, acc[t], false, false);
        }
    }
    float bn = bias[n];
#pragma unroll
    for (int t = 0; t < 4; ++t)
#pragma unroll
        for (int j = 0; j < 8; ++j) {
            int m = t * 16 + j + 8 * grp;
            out[(size_t)m * VV + n] = acc[t][j] + bn;
        }
}

// ---------------------------------------------------------------- log_softmax
__global__ void log_softmax_kernel(const float* __restrict__ logits,
                                   float* __restrict__ out) {
    int b = blockIdx.x, tid = threadIdx.x;
    __shared__ float red[256];
    const float* row = logits + (size_t)b * VV;
    float m = -3.4e38f;
    for (int i = tid; i < VV; i += 256) m = fmaxf(m, row[i]);
    red[tid] = m; __syncthreads();
    for (int s = 128; s > 0; s >>= 1) {
        if (tid < s) red[tid] = fmaxf(red[tid], red[tid + s]);
        __syncthreads();
    }
    m = red[0]; __syncthreads();
    float sum = 0.f;
    for (int i = tid; i < VV; i += 256) sum += __expf(row[i] - m);
    red[tid] = sum; __syncthreads();
    for (int s = 128; s > 0; s >>= 1) {
        if (tid < s) red[tid] += red[tid + s];
        __syncthreads();
    }
    float lse = m + __logf(red[0]);
    for (int i = tid; i < VV; i += 256)
        out[(size_t)b * VV + i] = row[i] - lse;
}

// =========================================================== host launcher
extern "C" void kernel_launch(void* const* d_in, const int* in_sizes, int n_in,
                              void* d_out, int out_size, void* d_ws, size_t ws_size,
                              hipStream_t stream) {
    const int*   word   = (const int*)  d_in[0];
    const float* h      = (const float*)d_in[1];   // (1,B,H) -> (B,H)
    const float* e_gpv  = (const float*)d_in[2];
    const float* e_am   = (const float*)d_in[3];
    const float* e_mt   = (const float*)d_in[4];
    const float* e_wx   = (const float*)d_in[5];
    const float* emb    = (const float*)d_in[6];
    const float* gW     = (const float*)d_in[7];
    const float* gb     = (const float*)d_in[8];
    const float* gv     = (const float*)d_in[9];
    const float* alW    = (const float*)d_in[10];
    const float* alb    = (const float*)d_in[11];
    const float* aW     = (const float*)d_in[12];
    const float* ab     = (const float*)d_in[13];
    const float* av     = (const float*)d_in[14];
    const float* mlW    = (const float*)d_in[15];
    const float* mlb    = (const float*)d_in[16];
    const float* mW     = (const float*)d_in[17];
    const float* mb     = (const float*)d_in[18];
    const float* mv     = (const float*)d_in[19];
    const float* wW     = (const float*)d_in[20];
    const float* wb     = (const float*)d_in[21];
    const float* wv     = (const float*)d_in[22];
    const float* cW     = (const float*)d_in[23];
    const float* cb     = (const float*)d_in[24];
    const float* Wih    = (const float*)d_in[25];
    const float* Whh    = (const float*)d_in[26];
    const float* bih    = (const float*)d_in[27];
    const float* bhh    = (const float*)d_in[28];
    const float* oW     = (const float*)d_in[29];
    const float* ob     = (const float*)d_in[30];

    float* out = (float*)d_out;
    float* out_logp = out;                         // 64*32000
    float* out_h    = out + (size_t)BB * VV;       // 64*1024
    float* out_we   = out_h + (size_t)BB * HH;     // 64*512

    char* ws = (char*)d_ws;
    float* cat     = (float*)(ws + 0);             // 64*3584
    float* h_am    = (float*)(ws + 917504);        // 64*512
    float* h_mt    = (float*)(ws + 1048576);
    float* hwb_g   = (float*)(ws + 1179648);       // 64*1024
    float* hwb_w   = (float*)(ws + 1441792);
    float* hwb_a   = (float*)(ws + 1703936);       // 64*512
    float* hwb_m   = (float*)(ws + 1835008);
    float* sc_g    = (float*)(ws + 1966080);       // 16384
    float* sc_a    = (float*)(ws + 2031616);
    float* sc_m    = (float*)(ws + 2097152);       // 4096
    float* sc_w    = (float*)(ws + 2113536);
    float* rnn_in  = (float*)(ws + 2179072);       // 64*1024
    float* gi      = (float*)(ws + 2441216);       // 64*3072
    float* gh      = (float*)(ws + 3227648);
    float* hnew    = (float*)(ws + 4014080);       // 64*1024
    float* logits  = (float*)(ws + 4276224);       // 64*32000
    __bf16* we_g   = (__bf16*)(ws + 12468224);     // 1024*1024
    __bf16* we_w   = (__bf16*)(ws + 14565376);
    __bf16* we_a   = (__bf16*)(ws + 16662528);     // 512*512
    __bf16* we_m   = (__bf16*)(ws + 17186816);

    auto cdiv = [](int a, int b) { return (a + b - 1) / b; };

    // 1) embedding gather (also fills cat[:, 0:512])
    embed_kernel<<<BB, 256, 0, stream>>>(word, emb, out_we, cat);

    // 2) h projections
    gemm_rowdot<<<cdiv(BB*HMM,256),256,0,stream>>>(h,HH, alW,HH,0, alb, h_am,HMM, BB,HMM,HH,0);
    gemm_rowdot<<<cdiv(BB*HMM,256),256,0,stream>>>(h,HH, mlW,HH,0, mlb, h_mt,HMM, BB,HMM,HH,0);

    // 3) t-independent energy terms hWb = bias + h_proj @ W_h^T
    gemm_rowdot<<<cdiv(BB*HH,256),256,0,stream>>>(h,HH,    gW,2*HH,0,  gb, hwb_g,HH,  BB,HH,HH,0);
    gemm_rowdot<<<cdiv(BB*HH,256),256,0,stream>>>(h,HH,    wW,2*HH,0,  wb, hwb_w,HH,  BB,HH,HH,0);
    gemm_rowdot<<<cdiv(BB*HMM,256),256,0,stream>>>(h_am,HMM, aW,2*HMM,0, ab, hwb_a,HMM, BB,HMM,HMM,0);
    gemm_rowdot<<<cdiv(BB*HMM,256),256,0,stream>>>(h_mt,HMM, mW,2*HMM,0, mb, hwb_m,HMM, BB,HMM,HMM,0);

    // 4) W_e slices -> bf16
    cvt_we_kernel<<<cdiv(HH*HH,256),256,0,stream>>>(gW,2*HH,HH,   we_g, HH,HH);
    cvt_we_kernel<<<cdiv(HH*HH,256),256,0,stream>>>(wW,2*HH,HH,   we_w, HH,HH);
    cvt_we_kernel<<<cdiv(HMM*HMM,256),256,0,stream>>>(aW,2*HMM,HMM, we_a, HMM,HMM);
    cvt_we_kernel<<<cdiv(HMM*HMM,256),256,0,stream>>>(mW,2*HMM,HMM, we_m, HMM,HMM);

    // 5) attention scores (bf16 WMMA, fp32 accumulate), 128-row tiles in LDS
    {
        size_t sm_h  = (size_t)MT * HH  * 2 + 8 * MT * 4;   // 266 KB (<320 KB WGP LDS)
        size_t sm_hm = (size_t)MT * HMM * 2 + 8 * MT * 4;   // 135 KB
        attn_score_kernel<HH> <<<256*BB/MT, 256, sm_h,  stream>>>(e_gpv, we_g, hwb_g, gv, sc_g);
        attn_score_kernel<HMM><<<256*BB/MT, 256, sm_hm, stream>>>(e_am,  we_a, hwb_a, av, sc_a);
        attn_score_kernel<HMM><<< 64*BB/MT, 256, sm_hm, stream>>>(e_mt,  we_m, hwb_m, mv, sc_m);
        attn_score_kernel<HH> <<<256*BB/MT, 256, sm_h,  stream>>>(e_wx,  we_w, hwb_w, wv, sc_w);
    }

    // 6) softmax + context, written into cat columns [we|gpv|am|mt|wx]
    softmax_ctx_kernel<<<BB,256,0,stream>>>(e_gpv, sc_g, cat + 512,  CATW, 256, HH);
    softmax_ctx_kernel<<<BB,256,0,stream>>>(e_am,  sc_a, cat + 1536, CATW, 256, HMM);
    softmax_ctx_kernel<<<BB,256,0,stream>>>(e_mt,  sc_m, cat + 2048, CATW,  64, HMM);
    softmax_ctx_kernel<<<BB,256,0,stream>>>(e_wx,  sc_w, cat + 2560, CATW, 256, HH);

    // 7) combine + relu
    gemm_rowdot<<<cdiv(BB*HH,256),256,0,stream>>>(cat,CATW, cW,CATW,0, cb, rnn_in,HH, BB,HH,CATW,1);

    // 8) GRU gates
    gemm_rowdot<<<cdiv(BB*3*HH,256),256,0,stream>>>(rnn_in,HH, Wih,HH,0, bih, gi,3*HH, BB,3*HH,HH,0);
    gemm_rowdot<<<cdiv(BB*3*HH,256),256,0,stream>>>(h,HH,      Whh,HH,0, bhh, gh,3*HH, BB,3*HH,HH,0);
    gru_gate_kernel<<<cdiv(BB*HH,256),256,0,stream>>>(gi, gh, h, hnew, out_h);

    // 9) vocab logits via fp32 WMMA, then log_softmax
    logits_wmma_kernel<<<VV/128, 256, 0, stream>>>(hnew, oW, ob, logits);
    log_softmax_kernel<<<BB, 256, 0, stream>>>(logits, out_logp);
}
